// LocalAggregator_31516470018770
// MI455X (gfx1250) — matrix operand order repaired and checked by reading, not compiled, not run on hardware
//
#include <hip/hip_runtime.h>
#include <hip/hip_bf16.h>

// GAT-style local aggregator for MI455X (gfx1250), wave32 + WMMA f16.
// Score GEMMs and the aggregation GEMM both run on v_wmma_f32_16x16x32_f16.

typedef __attribute__((ext_vector_type(16))) _Float16 v16h;
typedef __attribute__((ext_vector_type(8)))  _Float16 v8h;
typedef __attribute__((ext_vector_type(8)))  float    v8f;

#define B_   64
#define N_   512
#define D_   128
#define LRA_ 0.2f
#define NEGINF_ (-9.0e15f)

// ---- fragment loaders -------------------------------------------------------
// A-matrix f16 16x32 layout (ISA 7.12.2): lane<16 holds row M=lane,
// K = {k0..k0+7} in v0..v3 and {k0+16..k0+23} in v4..v7; lane>=16 holds
// row M=lane-16 with K offset by 8.  base must already include k0 + 8*khalf.
template <typename P>
__device__ __forceinline__ v16h load_afrag(P base) {
  v8h lo = *(const v8h*)(base);
  v8h hi = *(const v8h*)(base + 16);
  v16h r;
#pragma unroll
  for (int t = 0; t < 8; ++t) { r[t] = lo[t]; r[8 + t] = hi[t]; }
  return r;
}

// B-matrix f16 32x16 layout: lane<16 holds column N=lane with K = {k0..k0+7,
// k0+16..k0+23}, lane>=16 holds column N=lane-16 with K offset by 8.
// base must already include k0 + 8*khalf.
template <typename P>
__device__ __forceinline__ v16h load_bfrag(P base) {
  v8h lo = *(const v8h*)(base);
  v8h hi = *(const v8h*)(base + 16);
  v16h r;
#pragma unroll
  for (int t = 0; t < 8; ++t) { r[t] = lo[t]; r[8 + t] = hi[t]; }
  return r;
}

// ---- prep: h = [hidden[:,:,:127], p0*exp(-t)+p1] in f16, plus transpose ----
__global__ void prep_h(const float* __restrict__ hidden,
                       const float* __restrict__ times,
                       const float* __restrict__ p0,
                       const float* __restrict__ p1,
                       _Float16* __restrict__ hF,
                       _Float16* __restrict__ hT) {
  const int n = blockIdx.x, b = blockIdx.y, d = threadIdx.x;
  float v;
  if (d < D_ - 1) v = hidden[((size_t)(b * N_ + n)) * D_ + d];
  else            v = p0[0] * __expf(-times[b * N_ + n]) + p1[0];
  const _Float16 hv = (_Float16)v;
  hF[((size_t)(b * N_ + n)) * D_ + d] = hv;
  hT[((size_t)(b * D_ + d)) * N_ + n] = hv;
}

__global__ void prep_a(const float* __restrict__ a0, const float* __restrict__ a1,
                       const float* __restrict__ a2, const float* __restrict__ a3,
                       _Float16* __restrict__ a4) {
  const int t = threadIdx.x;               // 512 threads: 4 vectors x 128
  const int d = t & 127;
  float v = (t < 128) ? a0[d] : (t < 256) ? a1[d] : (t < 384) ? a2[d] : a3[d];
  a4[t] = (_Float16)v;
}

// ---- main: one 16-row tile per 256-thread block (8 waves) ------------------
__global__ __launch_bounds__(256)
void gat_kernel(const _Float16* __restrict__ hF,
                const _Float16* __restrict__ hT,
                const _Float16* __restrict__ a4,
                const int* __restrict__ adj,
                float* __restrict__ out) {
  const int b     = blockIdx.y;
  const int i0    = blockIdx.x * 16;
  const int tid   = threadIdx.x;
  const int w     = tid >> 5;
  const int lane  = tid & 31;
  const int l15   = lane & 15;
  const int khalf = lane >> 4;

  __shared__ _Float16 Pl[16][N_];   // softmaxed probabilities, 16 KB
  __shared__ float pmax[8][16];
  __shared__ float psum[8][16];

  // ---- pre-scaled A fragments: rows (h * a_k), all K = 128 -----------------
  v16h Asc[4][4];
  {
    const _Float16* hrow = hF + ((size_t)(b * N_ + i0 + l15)) * D_;
#pragma unroll
    for (int kk = 0; kk < 4; ++kk) {
      const int k0 = kk * 32;
      const v16h ab = load_afrag(hrow + k0 + khalf * 8);
#pragma unroll
      for (int ai = 0; ai < 4; ++ai) {
        const v16h av = load_afrag(a4 + ai * D_ + k0 + khalf * 8);
        Asc[ai][kk] = ab * av;      // v_pk_mul_f16
      }
    }
  }

  // ---- scores + adj mask for this wave's 4 column tiles --------------------
  float sc[4][8];
  float m[8];
#pragma unroll
  for (int v = 0; v < 8; ++v) m[v] = NEGINF_;

#pragma unroll
  for (int q = 0; q < 4; ++q) {
    const int j0 = (w * 4 + q) * 16;
    v16h Bf[4];
    const _Float16* hcol = hF + ((size_t)(b * N_ + j0 + l15)) * D_;
#pragma unroll
    for (int kk = 0; kk < 4; ++kk)
      Bf[kk] = load_bfrag(hcol + kk * 32 + khalf * 8);

    v8f e[4];
#pragma unroll
    for (int ai = 0; ai < 4; ++ai) {
      v8f c = {};
#pragma unroll
      for (int kk = 0; kk < 4; ++kk)
        c = __builtin_amdgcn_wmma_f32_16x16x32_f16(false, Asc[ai][kk],
                                                   false, Bf[kk],
                                                   (short)0, c, false, false);
      e[ai] = c;
    }

#pragma unroll
    for (int v = 0; v < 8; ++v) {
      const int i = i0 + v + 8 * khalf;        // C layout: M = v + 8*(lane>=16)
      const int j = j0 + l15;                  //           N = lane & 15
      const int av = adj[((size_t)(b * N_ + i)) * N_ + j];
      float r = NEGINF_;
      if (av == 1) r = e[0][v];
      if (av == 2) r = e[1][v];
      if (av == 3) r = e[2][v];
      if (av == 4) r = e[3][v];
      if (av >= 1 && av <= 4) r = (r > 0.f) ? r : LRA_ * r;  // leaky relu
      sc[q][v] = r;
      m[v] = fmaxf(m[v], r);
    }
  }

  // ---- softmax: reduce over N within 16-lane groups, then across waves -----
#pragma unroll
  for (int v = 0; v < 8; ++v) {
    float t = m[v];
#pragma unroll
    for (int off = 8; off >= 1; off >>= 1)
      t = fmaxf(t, __shfl_xor(t, off, 16));
    m[v] = t;
  }
  if (l15 == 0) {
#pragma unroll
    for (int v = 0; v < 8; ++v) pmax[w][v + 8 * khalf] = m[v];
  }
  __syncthreads();
  float rm[8];
#pragma unroll
  for (int v = 0; v < 8; ++v) {
    float t = pmax[0][v + 8 * khalf];
#pragma unroll
    for (int ww = 1; ww < 8; ++ww) t = fmaxf(t, pmax[ww][v + 8 * khalf]);
    rm[v] = t;
  }

  float s[8];
#pragma unroll
  for (int v = 0; v < 8; ++v) s[v] = 0.f;
#pragma unroll
  for (int q = 0; q < 4; ++q)
#pragma unroll
    for (int v = 0; v < 8; ++v) {
      const float epow = __expf(sc[q][v] - rm[v]);
      sc[q][v] = epow;
      s[v] += epow;
    }
#pragma unroll
  for (int v = 0; v < 8; ++v) {
    float t = s[v];
#pragma unroll
    for (int off = 8; off >= 1; off >>= 1)
      t += __shfl_xor(t, off, 16);
    s[v] = t;
  }
  if (l15 == 0) {
#pragma unroll
    for (int v = 0; v < 8; ++v) psum[w][v + 8 * khalf] = s[v];
  }
  __syncthreads();
  float rs[8];
#pragma unroll
  for (int v = 0; v < 8; ++v) {
    float t = 0.f;
#pragma unroll
    for (int ww = 0; ww < 8; ++ww) t += psum[ww][v + 8 * khalf];
    rs[v] = 1.0f / t;
  }

  // ---- stash probabilities to LDS in f16 for the aggregation GEMM ----------
#pragma unroll
  for (int q = 0; q < 4; ++q) {
    const int j0 = (w * 4 + q) * 16;
#pragma unroll
    for (int v = 0; v < 8; ++v)
      Pl[v + 8 * khalf][j0 + l15] = (_Float16)(sc[q][v] * rs[v]);
  }
  __syncthreads();

  // ---- out[16 x 128] = P[16 x 512] @ h[512 x 128]; wave w -> d cols 16w ----
  const int n0 = w * 16;
  v8f acc = {};
  const _Float16* htrow = hT + ((size_t)(b * D_ + n0 + l15)) * N_;
#pragma unroll 4
  for (int kk = 0; kk < 16; ++kk) {
    const int k0 = kk * 32;
    const v16h Af = load_afrag(&Pl[l15][k0 + khalf * 8]);     // ds_load_b128
    const v16h Bf = load_bfrag(htrow + k0 + khalf * 8);       // global b128
    acc = __builtin_amdgcn_wmma_f32_16x16x32_f16(false, Af, false, Bf,
                                                 (short)0, acc, false, false);
  }
#pragma unroll
  for (int v = 0; v < 8; ++v) {
    const int i = i0 + v + 8 * khalf;
    out[((size_t)(b * N_ + i)) * D_ + n0 + l15] = acc[v];
  }
}

extern "C" void kernel_launch(void* const* d_in, const int* in_sizes, int n_in,
                              void* d_out, int out_size, void* d_ws, size_t ws_size,
                              hipStream_t stream) {
  const float* hidden = (const float*)d_in[0];
  const int*   adj    = (const int*)  d_in[1];
  const float* times  = (const float*)d_in[2];
  const float* a0     = (const float*)d_in[3];
  const float* a1     = (const float*)d_in[4];
  const float* a2     = (const float*)d_in[5];
  const float* a3     = (const float*)d_in[6];
  const float* p0     = (const float*)d_in[7];
  const float* p1     = (const float*)d_in[8];
  float* out = (float*)d_out;

  _Float16* hF = (_Float16*)d_ws;                       // [B,N,128]  8.4 MB
  _Float16* hT = hF + (size_t)B_ * N_ * D_;             // [B,128,N]  8.4 MB
  _Float16* a4 = hT + (size_t)B_ * N_ * D_;             // [4,128]    1 KB

  prep_h<<<dim3(N_, B_), D_, 0, stream>>>(hidden, times, p0, p1, hF, hT);
  prep_a<<<1, 512, 0, stream>>>(a0, a1, a2, a3, a4);
  gat_kernel<<<dim3(N_ / 16, B_), 256, 0, stream>>>(hF, hT, a4, adj, out);
}
